// Patch_CoBi_Loss_36129264894393
// MI455X (gfx1250) — matrix operand order repaired and checked by reading, not compiled, not run on hardware
//
#include <hip/hip_runtime.h>
#include <hip/hip_bf16.h>

// ---- problem constants -----------------------------------------------------
#define NB     4      // batch
#define CHN    3      // channels
#define HH     67
#define WW     67
#define PATCH  5
#define HF     63     // H - PATCH + 1
#define WF     63
#define PP     3969   // HF*WF
#define PPAD   3984   // 249 * 16
#define KK     75     // 3*5*5 features
#define KPAD   96     // 3 * 32 (three K-steps of WMMA f16 16x16x32)
#define TILES  249    // PPAD / 16
#define WSP    0.1f
#define EPS_C  1e-5f

typedef __attribute__((ext_vector_type(16))) _Float16 v16h;
typedef __attribute__((ext_vector_type(8)))  _Float16 v8h;
typedef __attribute__((ext_vector_type(8)))  float    v8f;

// ---- workspace layout (bytes) ---------------------------------------------
#define OFF_MEAN   0u
#define FEATB      ((size_t)NB * PPAD * KPAD * 2)          // 3,059,712 B each
#define OFF_IF     512u
#define OFF_TF     (OFF_IF + FEATB)
#define RB         ((size_t)NB * PPAD * 4)                 // 63,744 B
#define R4B        ((size_t)NB * PPAD * 16)                // 254,976 B
#define OFF_RI     (OFF_TF + FEATB)
#define OFF_RT4    (OFF_RI + RB)
#define OFF_PART   (OFF_RT4 + R4B)                         // NB*TILES floats

// ---------------------------------------------------------------------------
// Kernel 1: meanT[c] = mean over (n, p) of target patch features
// ---------------------------------------------------------------------------
__global__ void mean_kernel(const float* __restrict__ target,
                            float* __restrict__ meanT) {
  const int c = blockIdx.x;                 // 0..74
  const int ch = c / 25, rem = c % 25, ky = rem / 5, kx = rem % 5;
  float s = 0.0f;
  for (int idx = threadIdx.x; idx < NB * PP; idx += blockDim.x) {
    const int n = idx / PP, p = idx % PP;
    const int y = p / WF, x = p % WF;
    s += target[(((size_t)n * CHN + ch) * HH + (y + ky)) * WW + (x + kx)];
  }
  __shared__ float red[256];
  red[threadIdx.x] = s;
  __syncthreads();
  for (int off = 128; off > 0; off >>= 1) {
    if ((int)threadIdx.x < off) red[threadIdx.x] += red[threadIdx.x + off];
    __syncthreads();
  }
  if (threadIdx.x == 0) meanT[c] = red[0] / (float)(NB * PP);
}

// ---------------------------------------------------------------------------
// Kernel 2: build centered f16 feature matrices [n][p][k] (k padded to 96,
// p padded to 3984, pads = 0). For I: f32 squared norm rI[n][p]. For T: a
// float4 table {rT_eff, y/62, x/62, 0} with the column-pad 1e30 baked into
// rT_eff for p >= P, so the main loop needs zero integer ops per tile.
// ---------------------------------------------------------------------------
__global__ void build_kernel(const float* __restrict__ outp,
                             const float* __restrict__ targ,
                             const float* __restrict__ meanT,
                             _Float16* __restrict__ If, _Float16* __restrict__ Tf,
                             float* __restrict__ rI, float4* __restrict__ rT4) {
  const int p = blockIdx.x;        // 0..PPAD-1
  const int n = blockIdx.y;        // 0..3
  const int which = blockIdx.z;    // 0 -> I(output), 1 -> T(target)
  const int k = threadIdx.x;       // blockDim.x == 96
  const float* src = which ? targ : outp;
  _Float16* dstF = which ? Tf : If;

  float v = 0.0f;
  if (p < PP && k < KK) {
    const int ch = k / 25, rem = k % 25, ky = rem / 5, kx = rem % 5;
    const int y = p / WF, x = p % WF;
    v = src[(((size_t)n * CHN + ch) * HH + (y + ky)) * WW + (x + kx)] - meanT[k];
  }
  dstF[((size_t)n * PPAD + p) * KPAD + k] = (_Float16)v;

  float sq = v * v;                      // norms in f32 (pre-quantization)
  for (int off = 16; off > 0; off >>= 1) sq += __shfl_xor(sq, off, 32);
  __shared__ float wred[3];
  if ((threadIdx.x & 31) == 0) wred[threadIdx.x >> 5] = sq;
  __syncthreads();
  if (threadIdx.x == 0) {
    const float r = wred[0] + wred[1] + wred[2];
    if (which) {
      float4 q;
      q.x = (p < PP) ? r : 1.0e30f;              // pad baked in
      q.y = (float)(p / WF) * (1.0f / 62.0f);
      q.z = (float)(p % WF) * (1.0f / 62.0f);
      q.w = 0.0f;
      rT4[(size_t)n * PPAD + p] = q;
    } else {
      rI[(size_t)n * PPAD + p] = r;
    }
  }
}

// ---------------------------------------------------------------------------
// Kernel 3: main WMMA kernel. One WG = one (n, 16-row stripe).
// Pass 1: WMMA 16x16 tiles over all j, comb -> LDS (16 x 3984 f32 = 249 KB,
//         LDS-resident thanks to CDNA5's 320KB/WGP), row-min tracked.
// Pass 2: re-read LDS, exp-sum per row, k_max = wmax/sum, partial -> ws.
// Per-tile j metadata comes from one b96/b128 load of the rT4 table; the
// prefetch index is clamped arithmetically so the loop has no divergent
// control flow at all.
// ---------------------------------------------------------------------------
__global__ __launch_bounds__(256) void cobi_main(
    const _Float16* __restrict__ If, const _Float16* __restrict__ Tf,
    const float* __restrict__ rI, const float4* __restrict__ rT4,
    float* __restrict__ partial) {
  extern __shared__ float lds[];
  float* comb    = lds;                  // 16 * PPAD
  float* wavemin = lds + 16 * PPAD;      // 8 * 16
  float* rowmin  = wavemin + 128;        // 16
  float* kpart   = rowmin + 16;          // 8

  const int tile_i = blockIdx.x;         // 0..TILES-1
  const int n      = blockIdx.y;
  const int i0     = tile_i * 16;
  const int lane   = threadIdx.x & 31;
  const int wave   = threadIdx.x >> 5;
  const int hi     = lane >> 4;          // which 16-lane half
  const int lo     = lane & 15;

  const _Float16* Ibase = If + ((size_t)n * PPAD + i0) * KPAD;
  const float*    rIrow = rI + (size_t)n * PPAD + i0;
  const float4*   qrow  = rT4 + (size_t)n * PPAD;
  const _Float16* Tn    = Tf + (size_t)n * PPAD * KPAD;

  // A fragments: lane(hi,lo) = row lo,
  // K = kb + hi*16 .. +15 (contiguous, matches B)
  v16h afrag[3];
#pragma unroll
  for (int ks = 0; ks < 3; ++ks) {
    const _Float16* ap = Ibase + (size_t)lo * KPAD + ks * 32;
    afrag[ks] = *(const v16h*)(ap + hi * 16);
  }

  // per-lane row metadata (rows m = r + 8*hi are fixed for the whole WG)
  float yi8[8], xi8[8], rI8[8];
#pragma unroll
  for (int r = 0; r < 8; ++r) {
    const int m = r + 8 * hi;
    const int i = i0 + m;
    const int ii = i < PP ? i : PP - 1;
    yi8[r] = (float)(ii / WF) * (1.0f / 62.0f);
    xi8[r] = (float)(ii % WF) * (1.0f / 62.0f);
    rI8[r] = rIrow[m];
  }

  float minv[8];
#pragma unroll
  for (int r = 0; r < 8; ++r) minv[r] = 3.0e38f;

  // ---- pass 1: WMMA over all column tiles (wave-strided; EXEC stays full)
  for (int tj = wave; tj < TILES; tj += 8) {
    const int j0 = tj * 16;
    const _Float16* Bbase = Tn + (size_t)j0 * KPAD;

    // clamped, branch-free prefetch of the next tile's B block
    const int tjn = (tj + 8 < TILES) ? (tj + 8) : tj;   // v_cndmask, no branch
    __builtin_prefetch(Tn + (size_t)tjn * 16 * KPAD, 0, 1);

    v8f c = {};
#pragma unroll
    for (int ks = 0; ks < 3; ++ks) {
      // B (32x16) striping: lane(hi,lo) = column lo, K = kb + hi*16 .. +15
      v16h b = *(const v16h*)(Bbase + (size_t)lo * KPAD + ks * 32 + hi * 16);
      c = __builtin_amdgcn_wmma_f32_16x16x32_f16(false, afrag[ks], false, b,
                                                 (short)0, c, false, false);
    }

    // C/D layout: lane col = lo, rows m = r + 8*hi.
    const int    j = j0 + lo;
    const float4 q = qrow[j];            // {rT_eff(+pad), yj, xj, 0}
    float* crow = comb + (8 * hi) * PPAD + j;
#pragma unroll
    for (int r = 0; r < 8; ++r) {
      const float dy = yi8[r] - q.y, dx = xi8[r] - q.z;
      const float sd = fmaf(dy, dy, dx * dx);
      float dist = fmaf(-2.0f, c[r], rI8[r] + q.x);
      dist = fmaxf(dist, 0.0f);
      const float cb = fmaf(WSP, sd, dist);
      crow[r * PPAD] = cb;
      minv[r] = fminf(cb, minv[r]);
    }
  }

  // min across the 16 columns held by each half-wave
#pragma unroll
  for (int r = 0; r < 8; ++r) {
    float m = minv[r];
#pragma unroll
    for (int off = 1; off < 16; off <<= 1)
      m = fminf(m, __shfl_xor(m, off, 32));
    minv[r] = m;
  }
  if (lo == 0) {
#pragma unroll
    for (int r = 0; r < 8; ++r) wavemin[wave * 16 + (r + 8 * hi)] = minv[r];
  }
  __syncthreads();
  if (threadIdx.x < 16) {
    float m = wavemin[threadIdx.x];
    for (int w = 1; w < 8; ++w) m = fminf(m, wavemin[w * 16 + threadIdx.x]);
    rowmin[threadIdx.x] = m;
  }
  __syncthreads();

  // ---- pass 2: exp-sum per row from LDS; k_max = exp((1-dmin*inv)*2)/sum
  float kacc = 0.0f;
#pragma unroll
  for (int rr = 0; rr < 2; ++rr) {
    const int m = 2 * wave + rr;
    const int i = i0 + m;
    const float dmin = rowmin[m];
    const float inv = 1.0f / (dmin + EPS_C);
    float s = 0.0f;
    for (int j = lane; j < PP; j += 32)
      s += __expf((1.0f - comb[m * PPAD + j] * inv) * 2.0f);
#pragma unroll
    for (int off = 16; off > 0; off >>= 1) s += __shfl_xor(s, off, 32);
    if (lane == 0 && i < PP) {
      const float wmax = __expf((1.0f - dmin * inv) * 2.0f);
      kacc += wmax / s;
    }
  }
  if (lane == 0) kpart[wave] = kacc;
  __syncthreads();
  if (threadIdx.x == 0) {
    float t = 0.0f;
    for (int w = 0; w < 8; ++w) t += kpart[w];   // ordered: deterministic
    partial[n * TILES + tile_i] = t;
  }
}

// ---------------------------------------------------------------------------
// Kernel 4: loss = sum_n -log( (1/P) * sum_tiles partial[n][t] )
// ---------------------------------------------------------------------------
__global__ void finalize_kernel(const float* __restrict__ partial,
                                float* __restrict__ out) {
  if (threadIdx.x == 0 && blockIdx.x == 0) {
    float loss = 0.0f;
    for (int n = 0; n < NB; ++n) {
      float s = 0.0f;
      for (int t = 0; t < TILES; ++t) s += partial[n * TILES + t];
      loss += -logf(s / (float)PP);
    }
    out[0] = loss;
  }
}

// ---------------------------------------------------------------------------
extern "C" void kernel_launch(void* const* d_in, const int* in_sizes, int n_in,
                              void* d_out, int out_size, void* d_ws, size_t ws_size,
                              hipStream_t stream) {
  (void)in_sizes; (void)n_in; (void)out_size; (void)ws_size;
  const float* output = (const float*)d_in[0];
  const float* target = (const float*)d_in[1];
  char* ws = (char*)d_ws;

  float*    meanT   = (float*)(ws + OFF_MEAN);
  _Float16* If      = (_Float16*)(ws + OFF_IF);
  _Float16* Tf      = (_Float16*)(ws + OFF_TF);
  float*    rI      = (float*)(ws + OFF_RI);
  float4*   rT4     = (float4*)(ws + OFF_RT4);
  float*    partial = (float*)(ws + OFF_PART);

  mean_kernel<<<KK, 256, 0, stream>>>(target, meanT);
  build_kernel<<<dim3(PPAD, NB, 2), 96, 0, stream>>>(output, target, meanT,
                                                     If, Tf, rI, rT4);

  const size_t smem = (size_t)16 * PPAD * sizeof(float) +
                      (128 + 16 + 8) * sizeof(float);   // ~249.6 KB < 320 KB/WGP
  (void)hipFuncSetAttribute((const void*)cobi_main,
                            hipFuncAttributeMaxDynamicSharedMemorySize,
                            (int)smem);
  cobi_main<<<dim3(TILES, NB), 256, smem, stream>>>(If, Tf, rI, rT4, partial);
  finalize_kernel<<<1, 32, 0, stream>>>(partial, (float*)d_out);
}
